// RecurrentRGCN_74397423501962
// MI455X (gfx1250) — matrix operand order repaired and checked by reading, not compiled, not run on hardware
//
#include <hip/hip_runtime.h>
#include <cstdint>
#include <cstddef>

// ---------------------------------------------------------------------------
// Types for CDNA5 WMMA (gfx1250, wave32)
// ---------------------------------------------------------------------------
typedef __bf16 bf16;
typedef __attribute__((ext_vector_type(16))) __bf16 v16bf;
typedef __attribute__((ext_vector_type(8)))  float  v8f;
typedef unsigned int u32x4 __attribute__((ext_vector_type(4)));

union Frag16 { v16bf v; u32x4 q[2]; };

// Problem constants (match the reference generator)
constexpr int H   = 256;     // hidden
constexpr int R   = 480;     // num relations
constexpr int NE  = 23000;   // num entities
constexpr int NEP = 23040;   // entities padded to 64
constexpr int NT  = 8192;    // triples

__device__ __forceinline__ float sigmoidf_(float x) { return 1.f / (1.f + __expf(-x)); }

// Opaque zero: defeats loop-invariant hoisting of weight-fragment loads across
// the GRU time recurrence (round-1 spill pattern) WITHOUT turning the global
// pointers into generic/flat pointers (round-2 flat_load pattern). Adding an
// unknown-but-zero element offset keeps the loads as global_load_b128 while
// making their addresses non-invariant per time step.
__device__ __forceinline__ int opaque_zero() {
    int x = 0;
    asm volatile("" : "+v"(x));
    return x;
}

__device__ __forceinline__ v8f zero8() {
    v8f z;
#pragma unroll
    for (int i = 0; i < 8; ++i) z[i] = 0.f;
    return z;
}

// A-fragment (16x32 bf16, M x K) from row-major [16][rowStride] storage.
// Contiguous k order (not the ISA table order; B matches): lane l holds row m=l&15; kb=(l>>4)*16; halves 0-7 = K0+kb+0..7,
// halves 8-15 = K0+kb+8..15. Two 16B loads.
__device__ __forceinline__ v16bf loadA16(const bf16* base, int rowStride, int lane, int K0) {
    const int m  = lane & 15;
    const int kb = (lane >> 4) << 4;
    const bf16* p = base + (size_t)m * rowStride + K0 + kb;
    Frag16 f;
    f.q[0] = *(const u32x4*)(p);
    f.q[1] = *(const u32x4*)(p + 8);
    return f.v;
}

// B-fragment (32x16 bf16, K x N) from TRANSPOSED [N][rowStride=K] storage:
// lane l holds column n=l&15; kb=(l>>4)*16; 16 contiguous bf16 = one 32B load.
__device__ __forceinline__ v16bf loadB16(const bf16* baseT, int rowStride, int lane, int N0, int K0) {
    const int n  = lane & 15;
    const int kb = (lane >> 4) << 4;
    const bf16* p = baseT + (size_t)(N0 + n) * rowStride + K0 + kb;
    return *(const v16bf*)(p);
}

__device__ __forceinline__ v8f wmma_bf16(v16bf a, v16bf b, v8f c) {
    return __builtin_amdgcn_wmma_f32_16x16x32_bf16(false, a, false, b, (short)0, c, false, false);
}

// ---------------------------------------------------------------------------
// Small generic kernels (per-relation precompute: negligible FLOPs)
// ---------------------------------------------------------------------------

// C[m,j] = alpha * (A[m,:] @ W[:,j] + b[j]) + (res ? res[m,j] : 0)
__global__ void lin_kernel(const float* __restrict__ A, const float* __restrict__ W,
                           const float* __restrict__ b, const float* __restrict__ res,
                           float alpha, float* __restrict__ C, int M, int K, int N) {
    int idx = blockIdx.x * 256 + threadIdx.x;
    if (idx >= M * N) return;
    int m = idx / N, j = idx % N;
    const float* ar = A + (size_t)m * K;
    float acc = 0.f;
    for (int k = 0; k < K; ++k) acc = fmaf(ar[k], W[(size_t)k * N + j], acc);
    acc = alpha * (acc + b[j]);
    if (res) acc += res[idx];
    C[idx] = acc;
}

// C[i,j] = A[i,:] @ B[j,:]  (A @ B^T)
__global__ void abt_kernel(const float* __restrict__ A, const float* __restrict__ B,
                           float* __restrict__ C, int M, int K, int N) {
    int idx = blockIdx.x * 256 + threadIdx.x;
    if (idx >= M * N) return;
    int i = idx / N, j = idx % N;
    const float* ar = A + (size_t)i * K;
    const float* br = B + (size_t)j * K;
    float acc = 0.f;
    for (int k = 0; k < K; ++k) acc = fmaf(ar[k], br[k], acc);
    C[idx] = acc;
}

// Per-relation GRU cell: h2R = GRU(x=mappedRel, h=phRel) given giR/ghR (biases included)
__global__ void h2r_kernel(const float* __restrict__ giR, const float* __restrict__ ghR,
                           const float* __restrict__ phRel, float* __restrict__ h2R) {
    int idx = blockIdx.x * 256 + threadIdx.x;
    if (idx >= R * H) return;
    int rel = idx / H, h = idx % H;
    int base = rel * 3 * H;
    float rg = sigmoidf_(giR[base + h] + ghR[base + h]);
    float zg = sigmoidf_(giR[base + H + h] + ghR[base + H + h]);
    float ng = tanhf(giR[base + 2 * H + h] + rg * ghR[base + 2 * H + h]);
    h2R[idx] = (1.f - zg) * ng + zg * phRel[idx];
}

// dst[j][i] = bf16(src[i][j])  : produce [N][M] transposed bf16 copies (WMMA B operands)
__global__ void transpose_bf16_kernel(const float* __restrict__ src, bf16* __restrict__ dst,
                                      int M, int N) {
    int idx = blockIdx.x * 256 + threadIdx.x;
    if (idx >= M * N) return;
    int i = idx / N, j = idx % N;
    dst[(size_t)j * M + i] = (bf16)src[idx];
}

// Ebf[e][h] = bf16(pre_emb[e,:] @ Walign[:,h] + balign[h]); zero pad rows e>=NE
__global__ void ealign_kernel(const float* __restrict__ pre_emb, const float* __restrict__ Walign,
                              const float* __restrict__ balign, bf16* __restrict__ Ebf) {
    int e = blockIdx.x, h = threadIdx.x;
    float v = 0.f;
    if (e < NE) {
        const float* pr = pre_emb + (size_t)e * H;
        float acc = 0.f;
        for (int k = 0; k < H; ++k) acc = fmaf(pr[k], Walign[(size_t)k * H + h], acc);
        v = acc + balign[h];
    }
    Ebf[(size_t)e * H + h] = (bf16)v;
}

// Abf[n][h] = bf16( (pre_emb[sub_idx[n],:]@Walign[:,h] + balign[h]) * h2R[rel_idx[n]][h] )
__global__ void avec_kernel(const float* __restrict__ pre_emb, const int* __restrict__ sub_idx,
                            const int* __restrict__ rel_idx, const float* __restrict__ Walign,
                            const float* __restrict__ balign, const float* __restrict__ h2R,
                            bf16* __restrict__ Abf) {
    int n = blockIdx.x, h = threadIdx.x;
    const float* pr = pre_emb + (size_t)sub_idx[n] * H;
    float acc = 0.f;
    for (int k = 0; k < H; ++k) acc = fmaf(pr[k], Walign[(size_t)k * H + h], acc);
    acc += balign[h];
    acc *= h2R[(size_t)rel_idx[n] * H + h];
    Abf[(size_t)n * H + h] = (bf16)acc;
}

// ---------------------------------------------------------------------------
// GRU scan: one workgroup (8 waves) owns 16 triples, loops T=10 steps.
// x = softmax(mask(hist*base)) @ mappedRel   (WMMA, K=480)
// gates = x@Wih + h@Whh (+biases)            (WMMA, K=256, N=768, n-gate split)
// h (bf16 WMMA operand) is double-buffered in LDS so the gates GEMM can be
// split into two low-register passes without a write-after-read hazard.
// ---------------------------------------------------------------------------
__global__ __launch_bounds__(256) void gru_scan_kernel(
        const float* __restrict__ partial, const int* __restrict__ rel_idx,
        const float* __restrict__ baseRR, const bf16* __restrict__ mapRelT,
        const bf16* __restrict__ WihT, const bf16* __restrict__ WhhT,
        const float* __restrict__ bih, const float* __restrict__ bhh,
        float* __restrict__ gruH, int T) {
    __shared__ __align__(32) bf16 attn_s[16 * R];    // 15 KB
    __shared__ __align__(32) bf16 xb_s[16 * H];      // 8 KB
    __shared__ __align__(32) bf16 hbuf_s[2][16 * H]; // 16 KB (double buffer)
    __shared__ float h_s[16 * H];                    // 16 KB (f32 state)
    __shared__ float red_s[16 * 16];                 // 1 KB
    __shared__ int   rel_s[16];

    const int tid  = threadIdx.x;
    const int lane = tid & 31;
    const int wave = tid >> 5;
    const int n0   = blockIdx.x * 16;

    if (tid < 16) rel_s[tid] = rel_idx[n0 + tid];
    for (int i = tid; i < 16 * H; i += 256) {
        h_s[i] = 0.f;
        hbuf_s[0][i] = (bf16)0.f;
        hbuf_s[1][i] = (bf16)0.f;
    }
    __syncthreads();

    const int row = tid >> 4, sub = tid & 15;        // 16 threads per triple row
    const int TS  = T * R;                           // partial row stride
    const int cn  = lane & 15;
    const int mAdd = (lane >= 16) ? 8 : 0;

    for (int t = 0; t < T; ++t) {
        // fresh opaque (zero) offset each step: keeps weight fragments as
        // in-loop GLOBAL loads (L0/L2 hot) instead of hoisted/spilled regs,
        // and preserves the global address space (no flat_load).
        const int zo = opaque_zero();
        const bf16* mapL = mapRelT + zo;
        const bf16* WihL = WihT + zo;
        const bf16* WhhL = WhhT + zo;
        const bf16* hb_cur = hbuf_s[t & 1];
        bf16*       hb_nxt = hbuf_s[(t + 1) & 1];

        const float* prow = partial + (size_t)(n0 + row) * TS + t * R;
        const float* brow = baseRR + (size_t)rel_s[row] * R;

        // ---- masked softmax over 480 relations (3 cheap passes, L0-hot) ----
        float lmax = -3.0e38f;
        for (int k = sub; k < R; k += 16) {
            float v = prow[k] * brow[k];
            v = (v == 0.f) ? -1e9f : v;
            lmax = fmaxf(lmax, v);
        }
        red_s[row * 16 + sub] = lmax;
        __syncthreads();
        float mx = -3.0e38f;
        for (int j = 0; j < 16; ++j) mx = fmaxf(mx, red_s[row * 16 + j]);
        __syncthreads();
        float lsum = 0.f;
        for (int k = sub; k < R; k += 16) {
            float v = prow[k] * brow[k];
            v = (v == 0.f) ? -1e9f : v;
            lsum += __expf(v - mx);
        }
        red_s[row * 16 + sub] = lsum;
        __syncthreads();
        float tot = 0.f;
        for (int j = 0; j < 16; ++j) tot += red_s[row * 16 + j];
        float inv = 1.f / tot;
        for (int k = sub; k < R; k += 16) {
            float v = prow[k] * brow[k];
            v = (v == 0.f) ? -1e9f : v;
            attn_s[row * R + k] = (bf16)(__expf(v - mx) * inv);
        }
        __syncthreads();

        // ---- x = attn[16,480] @ mappedRel[480,256] : wave -> 2 N-tiles ----
        {
            v8f cx0 = zero8(), cx1 = zero8();
#pragma unroll 1
            for (int kt = 0; kt < R / 32; ++kt) {
                int K0 = kt * 32;
                v16bf a = loadA16(attn_s, R, lane, K0);
                cx0 = wmma_bf16(a, loadB16(mapL, R, lane, (wave * 2 + 0) * 16, K0), cx0);
                cx1 = wmma_bf16(a, loadB16(mapL, R, lane, (wave * 2 + 1) * 16, K0), cx1);
            }
            {
                int col0 = (wave * 2 + 0) * 16 + cn;
                int col1 = (wave * 2 + 1) * 16 + cn;
#pragma unroll
                for (int v = 0; v < 8; ++v) {
                    xb_s[(v + mAdd) * H + col0] = (bf16)cx0[v];
                    xb_s[(v + mAdd) * H + col1] = (bf16)cx1[v];
                }
            }
        }
        __syncthreads();

        // ---- gates: two sequential N-tile passes, 4 accumulators each ----
#pragma unroll 1
        for (int j = 0; j < 2; ++j) {
            const int C0 = (wave * 2 + j) * 16;
            v8f cr = zero8(), cz = zero8(), cni = zero8(), cnh = zero8();
#pragma unroll 1
            for (int kt = 0; kt < H / 32; ++kt) {
                int K0 = kt * 32;
                v16bf ax = loadA16(xb_s, H, lane, K0);
                v16bf ah = loadA16(hb_cur, H, lane, K0);
                cr  = wmma_bf16(ax, loadB16(WihL, H, lane, C0,       K0), cr);
                cz  = wmma_bf16(ax, loadB16(WihL, H, lane, C0 + 256, K0), cz);
                cni = wmma_bf16(ax, loadB16(WihL, H, lane, C0 + 512, K0), cni);
                cr  = wmma_bf16(ah, loadB16(WhhL, H, lane, C0,       K0), cr);
                cz  = wmma_bf16(ah, loadB16(WhhL, H, lane, C0 + 256, K0), cz);
                cnh = wmma_bf16(ah, loadB16(WhhL, H, lane, C0 + 512, K0), cnh);
            }
            // GRU gate math + h update (writes only the NEXT h buffer)
            const int col = C0 + cn;
            float br_ = bih[col] + bhh[col];
            float bz_ = bih[col + 256] + bhh[col + 256];
            float bni = bih[col + 512];
            float bnh = bhh[col + 512];
#pragma unroll
            for (int v = 0; v < 8; ++v) {
                int m = v + mAdd;
                float rg = sigmoidf_(cr[v] + br_);
                float zg = sigmoidf_(cz[v] + bz_);
                float ng = tanhf((cni[v] + bni) + rg * (cnh[v] + bnh));
                float hp = h_s[m * H + col];
                float hn = (1.f - zg) * ng + zg * hp;
                h_s[m * H + col] = hn;
                hb_nxt[m * H + col] = (bf16)hn;
            }
        }
        __syncthreads();
    }

    for (int i = tid; i < 16 * H; i += 256)
        gruH[(size_t)(n0 + (i >> 8)) * H + (i & 255)] = h_s[i];
}

// ---------------------------------------------------------------------------
// Scoring GEMM + epilogue: C[8192,23000] = Abf @ Ebf^T (bf16 in, f32 accum).
// 64x64 tile per workgroup; A staged in LDS; B streamed from L2-resident Ebf.
// Writes sigmoid(score), accumulates softplus-sum & positive-sum per block.
// ---------------------------------------------------------------------------
__global__ __launch_bounds__(256) void score_kernel(
        const bf16* __restrict__ Abf, const bf16* __restrict__ Ebf,
        const int* __restrict__ obj_idx, float* __restrict__ score,
        float* __restrict__ spP, float* __restrict__ posP) {
    __shared__ __align__(32) bf16 As[64 * H];  // 32 KB
    __shared__ float red[256];

    const int tid  = threadIdx.x;
    const int lane = tid & 31;
    const int wave = tid >> 5;
    const int eBase = blockIdx.x * 64;
    const int mBase = blockIdx.y * 64;

    {   // contiguous 32 KB A-tile copy
        const u32x4* src = (const u32x4*)(Abf + (size_t)mBase * H);
        u32x4* dst = (u32x4*)As;
        for (int i = tid; i < 64 * H / 8; i += 256) dst[i] = src[i];
    }
    __syncthreads();

    const int mi = wave >> 1;
    const int niBase = (wave & 1) * 2;
    v8f acc0 = zero8(), acc1 = zero8();

#pragma unroll 1
    for (int kt = 0; kt < H / 32; ++kt) {
        int K0 = kt * 32;
        v16bf a = loadA16(As + (size_t)mi * 16 * H, H, lane, K0);
        acc0 = wmma_bf16(a, loadB16(Ebf, H, lane, eBase + (niBase + 0) * 16, K0), acc0);
        acc1 = wmma_bf16(a, loadB16(Ebf, H, lane, eBase + (niBase + 1) * 16, K0), acc1);
    }

    const int cn = lane & 15, mAdd = (lane >= 16) ? 8 : 0;
    float sp = 0.f, ps = 0.f;
#pragma unroll
    for (int j = 0; j < 2; ++j) {
        const v8f& acc = (j == 0) ? acc0 : acc1;
        int e = eBase + (niBase + j) * 16 + cn;
        if (e < NE) {
#pragma unroll
            for (int v = 0; v < 8; ++v) {
                int m = mBase + mi * 16 + v + mAdd;
                float s = acc[v];
                score[(size_t)m * NE + e] = sigmoidf_(s);
                sp += fmaxf(s, 0.f) + log1pf(__expf(-fabsf(s)));  // softplus(s)
                if (obj_idx[m] == e) ps += s;
            }
        }
    }

    red[tid] = sp; __syncthreads();
    for (int s = 128; s > 0; s >>= 1) { if (tid < s) red[tid] += red[tid + s]; __syncthreads(); }
    float spTot = red[0]; __syncthreads();
    red[tid] = ps; __syncthreads();
    for (int s = 128; s > 0; s >>= 1) { if (tid < s) red[tid] += red[tid + s]; __syncthreads(); }
    if (tid == 0) {
        int b = blockIdx.y * gridDim.x + blockIdx.x;
        spP[b] = spTot;
        posP[b] = red[0];
    }
}

// Per-triple partial of sum((predicted_hist - gru_hidden)^2)
__global__ void match_kernel(const float* __restrict__ phRel, const int* __restrict__ rel_idx,
                             const float* __restrict__ gruH, float* __restrict__ matchP) {
    __shared__ float red[256];
    int n = blockIdx.x, tid = threadIdx.x;
    float d = phRel[(size_t)rel_idx[n] * H + tid] - gruH[(size_t)n * H + tid];
    red[tid] = d * d; __syncthreads();
    for (int s = 128; s > 0; s >>= 1) { if (tid < s) red[tid] += red[tid + s]; __syncthreads(); }
    if (tid == 0) matchP[n] = red[0];
}

// Deterministic final reductions -> out[0]=match_loss, out[1]=path_loss
__global__ void finalize_kernel(const float* __restrict__ matchP, const float* __restrict__ spP,
                                const float* __restrict__ posP, int nBlocks,
                                float* __restrict__ out) {
    __shared__ double red[256];
    int tid = threadIdx.x;
    double a = 0.0;
    for (int i = tid; i < NT; i += 256) a += (double)matchP[i];
    red[tid] = a; __syncthreads();
    for (int s = 128; s > 0; s >>= 1) { if (tid < s) red[tid] += red[tid + s]; __syncthreads(); }
    if (tid == 0) out[0] = (float)(red[0] / ((double)NT * (double)H));
    __syncthreads();
    double b = 0.0;
    for (int i = tid; i < nBlocks; i += 256) b += (double)spP[i] - (double)posP[i];
    red[tid] = b; __syncthreads();
    for (int s = 128; s > 0; s >>= 1) { if (tid < s) red[tid] += red[tid + s]; __syncthreads(); }
    if (tid == 0) out[1] = (float)(red[0] / ((double)NT * (double)NE));
}

// ---------------------------------------------------------------------------
// Host orchestration
// ---------------------------------------------------------------------------
extern "C" void kernel_launch(void* const* d_in, const int* in_sizes, int n_in,
                              void* d_out, int out_size, void* d_ws, size_t ws_size,
                              hipStream_t stream) {
    const float* pre_emb = (const float*)d_in[0];
    const float* r_emb   = (const float*)d_in[1];
    const float* partial = (const float*)d_in[2];
    const int*   sub_idx = (const int*)d_in[3];
    const int*   rel_idx = (const int*)d_in[4];
    const int*   obj_idx = (const int*)d_in[5];
    const float* Wm1 = (const float*)d_in[7];   const float* bm1 = (const float*)d_in[8];
    const float* Wm2 = (const float*)d_in[9];   const float* bm2 = (const float*)d_in[10];
    const float* Wattn = (const float*)d_in[11]; const float* battn = (const float*)d_in[12];
    const float* Wh1 = (const float*)d_in[13];  const float* bh1 = (const float*)d_in[14];
    const float* Wh2 = (const float*)d_in[15];  const float* bh2 = (const float*)d_in[16];
    const float* Walign = (const float*)d_in[17]; const float* balign = (const float*)d_in[18];
    const float* Wih = (const float*)d_in[19];  const float* Whh = (const float*)d_in[20];
    const float* bih = (const float*)d_in[21];  const float* bhh = (const float*)d_in[22];

    int T = in_sizes[2] / (NT * R);   // == cur_ts == 10 in the reference setup
    if (T <= 0) T = 10;

    // workspace bump allocator (256B aligned)
    char* wsp = (char*)d_ws; size_t off = 0;
    auto alloc = [&](size_t bytes) -> char* {
        char* p = wsp + off;
        off += (bytes + 255) & ~(size_t)255;
        return p;
    };
    float* tmp1      = (float*)alloc((size_t)R * 512 * 4);
    float* mappedRel = (float*)alloc((size_t)R * H * 4);
    float* attnR     = (float*)alloc((size_t)R * H * 4);
    float* baseRR    = (float*)alloc((size_t)R * R * 4);
    float* t2        = (float*)alloc((size_t)R * H * 4);
    float* phRel     = (float*)alloc((size_t)R * H * 4);
    float* giR       = (float*)alloc((size_t)R * 3 * H * 4);
    float* ghR       = (float*)alloc((size_t)R * 3 * H * 4);
    float* h2R       = (float*)alloc((size_t)R * H * 4);
    bf16*  mapRelT   = (bf16*)alloc((size_t)H * R * 2);     // [H][R]
    bf16*  WihT      = (bf16*)alloc((size_t)3 * H * H * 2); // [768][256]
    bf16*  WhhT      = (bf16*)alloc((size_t)3 * H * H * 2);
    float* gruH      = (float*)alloc((size_t)NT * H * 4);
    bf16*  aVecBf    = (bf16*)alloc((size_t)NT * H * 2);
    bf16*  Ebf       = (bf16*)alloc((size_t)NEP * H * 2);
    const int nScoreBlocks = (NEP / 64) * (NT / 64);        // 360*128 = 46080
    float* matchP    = (float*)alloc((size_t)NT * 4);
    float* spP       = (float*)alloc((size_t)nScoreBlocks * 4);
    float* posP      = (float*)alloc((size_t)nScoreBlocks * 4);

    float* out = (float*)d_out;
    float* scoreOut = out + 2;

    auto lgrid = [](int total) { return dim3((total + 255) / 256); };

    // --- per-relation precompute chain (all tiny) ---
    lin_kernel<<<lgrid(R * 512), 256, 0, stream>>>(r_emb, Wm1, bm1, nullptr, 1.f, tmp1, R, H, 512);
    lin_kernel<<<lgrid(R * H), 256, 0, stream>>>(tmp1, Wm2, bm2, nullptr, 1.f, mappedRel, R, 512, H);
    lin_kernel<<<lgrid(R * H), 256, 0, stream>>>(mappedRel, Wattn, battn, nullptr, 1.f, attnR, R, H, H);
    abt_kernel<<<lgrid(R * R), 256, 0, stream>>>(attnR, mappedRel, baseRR, R, H, R);
    lin_kernel<<<lgrid(R * H), 256, 0, stream>>>(mappedRel, Wh1, bh1, nullptr, 1.f, t2, R, H, H);
    lin_kernel<<<lgrid(R * H), 256, 0, stream>>>(t2, Wh2, bh2, mappedRel, 0.1f, phRel, R, H, H);
    lin_kernel<<<lgrid(R * 3 * H), 256, 0, stream>>>(mappedRel, Wih, bih, nullptr, 1.f, giR, R, H, 3 * H);
    lin_kernel<<<lgrid(R * 3 * H), 256, 0, stream>>>(phRel, Whh, bhh, nullptr, 1.f, ghR, R, H, 3 * H);
    h2r_kernel<<<lgrid(R * H), 256, 0, stream>>>(giR, ghR, phRel, h2R);

    // --- bf16 transposed operands for WMMA B ---
    transpose_bf16_kernel<<<lgrid(R * H), 256, 0, stream>>>(mappedRel, mapRelT, R, H);
    transpose_bf16_kernel<<<lgrid(H * 3 * H), 256, 0, stream>>>(Wih, WihT, H, 3 * H);
    transpose_bf16_kernel<<<lgrid(H * 3 * H), 256, 0, stream>>>(Whh, WhhT, H, 3 * H);

    // --- GRU scan over T steps (WMMA-heavy) ---
    gru_scan_kernel<<<NT / 16, 256, 0, stream>>>(partial, rel_idx, baseRR, mapRelT,
                                                 WihT, WhhT, bih, bhh, gruH, T);

    // --- match loss partials ---
    match_kernel<<<NT, 256, 0, stream>>>(phRel, rel_idx, gruH, matchP);

    // --- scoring operands ---
    ealign_kernel<<<NEP, 256, 0, stream>>>(pre_emb, Walign, balign, Ebf);
    avec_kernel<<<NT, 256, 0, stream>>>(pre_emb, sub_idx, rel_idx, Walign, balign, h2R, aVecBf);

    // --- big GEMM + sigmoid + loss partials (store-bandwidth bound) ---
    dim3 sg(NEP / 64, NT / 64);
    score_kernel<<<sg, 256, 0, stream>>>(aVecBf, Ebf, obj_idx, scoreOut, spP, posP);

    // --- deterministic final reduction ---
    finalize_kernel<<<1, 256, 0, stream>>>(matchP, spP, posP, nScoreBlocks, out);

    (void)n_in; (void)out_size; (void)ws_size; (void)in_sizes;
}
